// CapsuleLayer_84293028152110
// MI455X (gfx1250) — matrix operand order repaired and checked by reading, not compiled, not run on hardware
//
#include <hip/hip_runtime.h>

typedef __attribute__((ext_vector_type(16))) _Float16 v16h;
typedef __attribute__((ext_vector_type(2)))  _Float16 h2;
typedef __attribute__((ext_vector_type(8)))  float    v8f;

#define BATCH 512
#define RDIM  640
#define CDIM  10
#define ODIM  16
#define IDIM  8
#define ITERS 3
#define EPSV  1e-8f
#define BT    16                 // batch tile = WMMA N (lanes)
#define NW    CDIM               // 10 waves per block
#define NTHREADS (NW * 32)
#define RPW   (RDIM / NW)        // 64 r's per wave

#define X_BYTES    (RDIM * BT * IDIM * 2)       // 160 KB packed f16 x-tile
#define S_BYTES    (NW * CDIM * 32 * 8 * 4)     // 100 KB partial s_j
#define V_BYTES    (CDIM * 32 * 8 * 4)          // 10 KB v_j broadcast
#define SMEM_BYTES (X_BYTES + S_BYTES + V_BYTES)

// W[r,c,o=l16,0..7] as WMMA A operand rows (elements 0..3 = K 0..7 as h2).
// No exec guard: lanes 16-31 feed K=8..15, which our B operand zeroes, so
// their (finite) values are multiplied by 0.
__device__ __forceinline__ v16h load_w_f16(const float* __restrict__ W,
                                           int r, int c, int o) {
    const float4* wp = (const float4*)(W + (((size_t)r * CDIM + c) * ODIM + o) * IDIM);
    float4 w0 = wp[0], w1 = wp[1];
    v16h a = {};
    ((h2*)&a)[0] = __builtin_bit_cast(h2, __builtin_amdgcn_cvt_pkrtz(w0.x, w0.y));
    ((h2*)&a)[1] = __builtin_bit_cast(h2, __builtin_amdgcn_cvt_pkrtz(w0.z, w0.w));
    ((h2*)&a)[2] = __builtin_bit_cast(h2, __builtin_amdgcn_cvt_pkrtz(w1.x, w1.y));
    ((h2*)&a)[3] = __builtin_bit_cast(h2, __builtin_amdgcn_cvt_pkrtz(w1.z, w1.w));
    return a;
}

// Wave w owns r in [w*64, w*64+64) and ALL capsules -> softmax in-register.
// D[o,b]: lane = b, VGPR v (+8*half) = o.  b_ij never materialized:
// b_ij = <u_hat, vsum>.  Pass 2 packs TWO r's along WMMA K and accumulates
// straight into the C operand: s[c] = wmma([W_r0|W_r1], [x_r0*cij0|x_r1*cij1], s[c]).
__global__ __launch_bounds__(NTHREADS, 1)
void capsule_routing_fused(const float* __restrict__ x,
                           const float* __restrict__ W,
                           const float* __restrict__ bias,
                           float* __restrict__ out)
{
    extern __shared__ char smem[];
    _Float16* lds_x = (_Float16*)smem;                      // [r][b] -> 8 halves
    float*    lds_s = (float*)(smem + X_BYTES);             // [w][c][lane][v0..7]
    float*    lds_v = (float*)(smem + X_BYTES + S_BYTES);   // [c][lane][v0..7]

    const int tid  = threadIdx.x;
    const int lane = tid & 31;
    const int w    = tid >> 5;          // wave id: r-block owner & squash capsule
    const int b0   = blockIdx.x * BT;
    const int half = lane >> 4;         // o-half
    const int l16  = lane & 15;         // b index
    const int rbeg = w * RPW;

    // ---- stage x tile (f32 -> f16, WMMA B-operand rows) ----
    for (int p = tid; p < RDIM * BT; p += NTHREADS) {
        const int r = p >> 4, b = p & 15;
        const float4* xp = (const float4*)(x + ((size_t)(b0 + b) * RDIM + r) * IDIM);
        float4 x0 = xp[0], x1 = xp[1];
        h2 q[4];
        q[0] = __builtin_bit_cast(h2, __builtin_amdgcn_cvt_pkrtz(x0.x, x0.y));
        q[1] = __builtin_bit_cast(h2, __builtin_amdgcn_cvt_pkrtz(x0.z, x0.w));
        q[2] = __builtin_bit_cast(h2, __builtin_amdgcn_cvt_pkrtz(x1.x, x1.y));
        q[3] = __builtin_bit_cast(h2, __builtin_amdgcn_cvt_pkrtz(x1.z, x1.w));
        ((uint4*)lds_x)[p] = *(uint4*)q;                    // ds_store_b128
    }
    __syncthreads();

    v8f bo;                                                 // bias[c=w, o=v+8*half]
#pragma unroll
    for (int v = 0; v < 8; ++v) bo[v] = bias[w * ODIM + v + 8 * half];

    const v8f zf = {};
    v8f vsum[CDIM];
#pragma unroll
    for (int c = 0; c < CDIM; ++c) vsum[c] = zf;
    v8f vout = zf;

    for (int it = 0; it < ITERS; ++it) {
        v8f s[CDIM];
#pragma unroll
        for (int c = 0; c < CDIM; ++c) s[c] = zf;

        for (int rl = 0; rl < RPW; rl += 2) {
            const int r2 = rbeg + rl;                       // r-pair base
            if (rl + 2 < RPW)                               // global_prefetch_b8
                __builtin_prefetch(W + (((size_t)(r2 + 2) * CDIM) * ODIM) * IDIM, 0, 1);

            // B operands: x rows for r2, r2+1 (guarded: upper lanes must be 0)
            v16h bm[2];
            h2  cpk[CDIM] = {};                             // {cij(r0), cij(r1)} per c
#pragma unroll
            for (int rr = 0; rr < 2; ++rr) {
                bm[rr] = (v16h){};
                if (lane < 16) {
                    uint4 t = ((const uint4*)lds_x)[(r2 + rr) * BT + l16];
                    *(uint4*)&bm[rr] = t;
                }
            }

            // ---- pass 1: per-r logits for all capsules, in-register softmax ----
#pragma unroll
            for (int rr = 0; rr < 2; ++rr) {
                const int r = r2 + rr;
                float lg[CDIM];
#pragma unroll
                for (int c = 0; c < CDIM; ++c) {
                    v16h amat = load_w_f16(W, r, c, l16);
                    v8f d = __builtin_amdgcn_wmma_f32_16x16x32_f16(
                        false, amat, false, bm[rr], (short)0, zf, false, false);
                    float t = 0.f;
#pragma unroll
                    for (int v = 0; v < 8; ++v) t = fmaf(d[v], vsum[c][v], t);
                    t += __shfl_xor(t, 16, 32);             // combine o halves
                    lg[c] = t;
                }
                float mx = lg[0];
#pragma unroll
                for (int c = 1; c < CDIM; ++c) mx = fmaxf(mx, lg[c]);
                float den = 0.f;
#pragma unroll
                for (int c = 0; c < CDIM; ++c) { lg[c] = __expf(lg[c] - mx); den += lg[c]; }
                const float rden = __builtin_amdgcn_rcpf(den);
#pragma unroll
                for (int c = 0; c < CDIM; ++c) cpk[c][rr] = (_Float16)(lg[c] * rden);
            }

            // ---- pass 2: 2-r K-packed accumulate into WMMA C operand ----
#pragma unroll
            for (int c = 0; c < CDIM; ++c) {
                h2 csA; csA[0] = cpk[c][0]; csA[1] = cpk[c][0];
                h2 csB; csB[0] = cpk[c][1]; csB[1] = cpk[c][1];
                v16h bs;
                h2* b0p = (h2*)&bm[0];
                h2* b1p = (h2*)&bm[1];
                h2* bsp = (h2*)&bs;
#pragma unroll
                for (int j = 0; j < 4; ++j) {               // v_pk_mul_f16
                    bsp[j]     = b0p[j] * csA;              // K 0..7  = x_r0 * cij0
                    bsp[j + 4] = b1p[j] * csB;              // K 8..15 = x_r1 * cij1
                }
                // A: elements 0..7 = W_r0 (K0..7), elements 8..15 = W_r1 (K8..15)
                v16h amat = load_w_f16(W, r2, c, l16);
                {
                    v16h a1 = load_w_f16(W, r2 + 1, c, l16);
#pragma unroll
                    for (int j = 0; j < 4; ++j) ((h2*)&amat)[j + 4] = ((h2*)&a1)[j];
                }
                s[c] = __builtin_amdgcn_wmma_f32_16x16x32_f16(
                    false, amat, false, bs, (short)0, s[c], false, false);
            }
        }

        // ---- cross-wave reduction of partial s (2 barriers per iteration) ----
#pragma unroll
        for (int c = 0; c < CDIM; ++c) {
            float* sp = lds_s + ((size_t)((w * CDIM + c) * 32 + lane) * 8);
            ((float4*)sp)[0] = make_float4(s[c][0], s[c][1], s[c][2], s[c][3]);
            ((float4*)sp)[1] = make_float4(s[c][4], s[c][5], s[c][6], s[c][7]);
        }
        __syncthreads();

        // wave w reduces & squashes its own capsule c == w
        v8f st = zf;
#pragma unroll
        for (int ww = 0; ww < NW; ++ww) {
            const float4* sp =
                (const float4*)(lds_s + ((size_t)((ww * CDIM + w) * 32 + lane) * 8));
            float4 a = sp[0], b = sp[1];
            st[0] += a.x; st[1] += a.y; st[2] += a.z; st[3] += a.w;
            st[4] += b.x; st[5] += b.y; st[6] += b.z; st[7] += b.w;
        }
#pragma unroll
        for (int v = 0; v < 8; ++v) st[v] += bo[v];
        float n2 = 0.f;
#pragma unroll
        for (int v = 0; v < 8; ++v) n2 = fmaf(st[v], st[v], n2);
        n2 += __shfl_xor(n2, 16, 32);                       // combine o halves
        const float nrm   = sqrtf(n2);
        const float scale = nrm / (1.f + n2 + EPSV);
#pragma unroll
        for (int v = 0; v < 8; ++v) vout[v] = scale * st[v];

        // publish v_j so every wave can update its full vsum[0..9]
        {
            float* vp = lds_v + ((size_t)(w * 32 + lane) * 8);
            ((float4*)vp)[0] = make_float4(vout[0], vout[1], vout[2], vout[3]);
            ((float4*)vp)[1] = make_float4(vout[4], vout[5], vout[6], vout[7]);
        }
        __syncthreads();
        if (it + 1 < ITERS) {
#pragma unroll
            for (int c = 0; c < CDIM; ++c) {
                const float4* vp = (const float4*)(lds_v + ((size_t)(c * 32 + lane) * 8));
                float4 a = vp[0], b = vp[1];
                vsum[c][0] += a.x; vsum[c][1] += a.y; vsum[c][2] += a.z; vsum[c][3] += a.w;
                vsum[c][4] += b.x; vsum[c][5] += b.y; vsum[c][6] += b.z; vsum[c][7] += b.w;
            }
        }
    }

    // ---- out[b, c=w, o]: lane b writes o = 8*half .. 8*half+7 ----
    float* op = out + (size_t)(b0 + l16) * (CDIM * ODIM) + w * ODIM + 8 * half;
    ((float4*)op)[0] = make_float4(vout[0], vout[1], vout[2], vout[3]);
    ((float4*)op)[1] = make_float4(vout[4], vout[5], vout[6], vout[7]);
}

extern "C" void kernel_launch(void* const* d_in, const int* in_sizes, int n_in,
                              void* d_out, int out_size, void* d_ws, size_t ws_size,
                              hipStream_t stream) {
    (void)in_sizes; (void)n_in; (void)d_ws; (void)ws_size; (void)out_size;
    const float* x    = (const float*)d_in[0];   // [512, 640, 8]
    const float* W    = (const float*)d_in[1];   // [640, 10, 16, 8]
    const float* bias = (const float*)d_in[2];   // [1, 1, 10, 16]
    float*       out  = (float*)d_out;           // [512, 10, 16]

    dim3 grid(BATCH / BT);          // 32 workgroups
    dim3 block(NTHREADS);           // 10 waves of 32
    capsule_routing_fused<<<grid, block, SMEM_BYTES, stream>>>(x, W, bias, out);
}
